// MPNNGNN_29326036697883
// MI455X (gfx1250) — matrix operand order repaired and checked by reading, not compiled, not run on hardware
//
#include <hip/hip_runtime.h>
#include <hip/hip_bf16.h>

// ---------------- problem sizes (fixed by the reference) ----------------
#define V_NODES 20000
#define E_EDGES 100000
#define NODE_IN 128
#define EDGE_IN 128
#define HDIM    64
#define EDGE_HID 128
#define OUT_F   64
#define N_STEPS 9

// ---------------- WMMA types / helpers (CDNA5 wave32) ----------------
typedef __attribute__((ext_vector_type(16))) __bf16 v16bf;
typedef __attribute__((ext_vector_type(8)))  float  v8f;

static __device__ __forceinline__ v8f wmma_bf16(v16bf a, v16bf b, v8f c) {
  // D(16x16,f32) = A(16x32,bf16) * B(32x16,bf16) + C
  return __builtin_amdgcn_wmma_f32_16x16x32_bf16(false, a, false, b,
                                                 (short)0, c, false, false);
}

static __device__ __forceinline__ unsigned short f_to_bf16u(float f) {
  unsigned int u = __float_as_uint(f);
  u += 0x7fffu + ((u >> 16) & 1u);          // round-to-nearest-even
  return (unsigned short)(u >> 16);
}

// A fragment (16x32) from row-major f32, rows start at ptr, K offset k0.
// Layout: lane<16 -> M=lane, K k0+0..15; lane>=16 -> K k0+16..31
static __device__ __forceinline__ v16bf load_a_f32(const float* __restrict__ A,
                                                   int lda, int k0) {
  int lane = threadIdx.x & 31;
  int m = lane & 15;
  int kh = (lane >> 4) << 4;
  const float* row = A + (size_t)m * lda + k0 + kh;
  v16bf a;
#pragma unroll
  for (int i = 0; i < 8; ++i) a[i] = (__bf16)row[i];
#pragma unroll
  for (int i = 0; i < 8; ++i) a[8 + i] = (__bf16)row[8 + i];
  return a;
}

// A fragment (16x32) from row-major bf16 (raw ushort bits)
static __device__ __forceinline__ v16bf load_a_bf16(const unsigned short* __restrict__ A,
                                                    int lda, int k0) {
  int lane = threadIdx.x & 31;
  int m = lane & 15;
  int kh = (lane >> 4) << 4;
  const unsigned short* row = A + (size_t)m * lda + k0 + kh;
  v16bf a;
#pragma unroll
  for (int i = 0; i < 8; ++i) a[i] = __builtin_bit_cast(__bf16, row[i]);
#pragma unroll
  for (int i = 0; i < 8; ++i) a[8 + i] = __builtin_bit_cast(__bf16, row[8 + i]);
  return a;
}

// B fragment (32x16) where B = W^T, W row-major (n_out, K): B[k][n] = W[n*ldw+k]
// lane<16 -> column n0+lane, K k0..k0+15 ; lane>=16 -> K k0+16..k0+31
static __device__ __forceinline__ v16bf load_b_wT_f32(const float* __restrict__ W,
                                                      int ldw, int k0, int n0) {
  int lane = threadIdx.x & 31;
  int n = lane & 15;
  int kh = (lane >> 4) << 4;
  const float* col = W + (size_t)(n0 + n) * ldw + k0 + kh;
  v16bf b;
#pragma unroll
  for (int i = 0; i < 16; ++i) b[i] = (__bf16)col[i];
  return b;
}

// B fragment from pre-packed fragment-major bf16 buffer:
// frag = ntile*(K/32) + ktile; layout: P[frag*512 + lane*16 + i]
// -> two fully coalesced global_load_b128 per fragment, zero cvt VALU.
static __device__ __forceinline__ v16bf load_b_packed(const unsigned short* __restrict__ P,
                                                      int frag) {
  int lane = threadIdx.x & 31;
  const unsigned short* p = P + (size_t)frag * 512 + lane * 16;
  v16bf b;
#pragma unroll
  for (int i = 0; i < 16; ++i) b[i] = __builtin_bit_cast(__bf16, p[i]);
  return b;
}

// ---------------- repack kernel: W (N x K, f32 row-major) -> bf16 B fragments
// one wave per (ntile, ktile) fragment; grid.x = (N/16)*(K/32)
__global__ void pack_b_kernel(const float* __restrict__ W, int K,
                              unsigned short* __restrict__ P) {
  int frag = blockIdx.x;
  int KT = K >> 5;
  int ntile = frag / KT;
  int kt = frag - ntile * KT;
  int lane = threadIdx.x & 31;
  int n = lane & 15;
  int kh = (lane >> 4) << 4;
  const float* col = W + (size_t)(ntile * 16 + n) * K + kt * 32 + kh;
  unsigned short* p = P + (size_t)frag * 512 + lane * 16;
#pragma unroll
  for (int i = 0; i < 16; ++i) p[i] = f_to_bf16u(col[i]);
}

// ---------------- kernel 1/6: fused Linear->ReLU->Linear (rows x KIN -> 64 -> 64)
template <int KIN>
__global__ void fused2layer_kernel(const float* __restrict__ X,
                                   const float* __restrict__ W1, const float* __restrict__ b1,
                                   const float* __restrict__ W2, const float* __restrict__ b2,
                                   float* __restrict__ Y) {
  __shared__ unsigned short tile[16 * 64];    // relu(layer1) staged as bf16
  int r0 = blockIdx.x * 16;
  int lane = threadIdx.x & 31;
  int n = lane & 15;
  int mb = (lane >> 4) * 8;

  // ---- layer 1: (16 x KIN) @ (KIN x 64) ----
#pragma unroll
  for (int nt = 0; nt < 4; ++nt) {
    int n0 = nt * 16;
    v8f c;
    float bv = b1[n0 + n];
#pragma unroll
    for (int i = 0; i < 8; ++i) c[i] = bv;
#pragma unroll
    for (int kt = 0; kt < KIN / 32; ++kt) {
      v16bf a = load_a_f32(X + (size_t)r0 * KIN, KIN, kt * 32);
      v16bf b = load_b_wT_f32(W1, KIN, kt * 32, n0);
      c = wmma_bf16(a, b, c);
    }
#pragma unroll
    for (int i = 0; i < 8; ++i)
      tile[(mb + i) * 64 + n0 + n] = f_to_bf16u(fmaxf(c[i], 0.0f));
  }
  __syncthreads();

  // ---- layer 2: (16 x 64) @ (64 x 64) ----
#pragma unroll
  for (int nt = 0; nt < 4; ++nt) {
    int n0 = nt * 16;
    v8f c;
    float bv = b2[n0 + n];
#pragma unroll
    for (int i = 0; i < 8; ++i) c[i] = bv;
#pragma unroll
    for (int kt = 0; kt < 2; ++kt) {
      v16bf a = load_a_bf16(tile, 64, kt * 32);
      v16bf b = load_b_wT_f32(W2, 64, kt * 32, n0);
      c = wmma_bf16(a, b, c);
    }
#pragma unroll
    for (int i = 0; i < 8; ++i)
      Y[(size_t)(r0 + mb + i) * 64 + n0 + n] = c[i];
  }
}

// ---------------- kernel 2: He = relu(Ef @ We1^T)  (E x 128) -> bf16
__global__ void edge_hidden_kernel(const float* __restrict__ Ef,
                                   const unsigned short* __restrict__ P1,
                                   const float* __restrict__ b1,
                                   unsigned short* __restrict__ He) {
  int r0 = blockIdx.x * 16;
  int n0 = blockIdx.y * 64;
  int lane = threadIdx.x & 31;
  int n = lane & 15;
  int mb = (lane >> 4) * 8;

  v16bf a[4];
#pragma unroll
  for (int kt = 0; kt < 4; ++kt)
    a[kt] = load_a_f32(Ef + (size_t)r0 * EDGE_IN, EDGE_IN, kt * 32);

#pragma unroll
  for (int nt = 0; nt < 4; ++nt) {
    int nn = n0 + nt * 16;
    v8f c;
    float bv = b1[nn + n];
#pragma unroll
    for (int i = 0; i < 8; ++i) c[i] = bv;
#pragma unroll
    for (int kt = 0; kt < 4; ++kt)
      c = wmma_bf16(a[kt], load_b_packed(P1, (nn >> 4) * 4 + kt), c);
#pragma unroll
    for (int i = 0; i < 8; ++i)
      He[(size_t)(r0 + mb + i) * EDGE_HID + nn + n] = f_to_bf16u(fmaxf(c[i], 0.0f));
  }
}

// ---------------- kernel 3: We = He @ We2^T + b2  (E x 4096) -> bf16 (big GEMM)
// wave computes 16 rows x 512 cols: A loaded once, 8 chunks of 64 cols.
__global__ void edge_we_kernel(const unsigned short* __restrict__ He,
                               const unsigned short* __restrict__ P2,
                               const float* __restrict__ b2,
                               unsigned short* __restrict__ We) {
  int r0 = blockIdx.x * 16;
  int cbase = blockIdx.y * 512;
  int lane = threadIdx.x & 31;
  int n = lane & 15;
  int mb = (lane >> 4) * 8;

  v16bf a[4];
#pragma unroll
  for (int kt = 0; kt < 4; ++kt)
    a[kt] = load_a_bf16(He + (size_t)r0 * EDGE_HID, EDGE_HID, kt * 32);

  for (int cc = 0; cc < 8; ++cc) {
    int ncc = cbase + cc * 64;
#pragma unroll
    for (int nt = 0; nt < 4; ++nt) {
      int nn = ncc + nt * 16;
      v8f c;
      float bv = b2[nn + n];
#pragma unroll
      for (int i = 0; i < 8; ++i) c[i] = bv;
#pragma unroll
      for (int kt = 0; kt < 4; ++kt)
        c = wmma_bf16(a[kt], load_b_packed(P2, (nn >> 4) * 4 + kt), c);
#pragma unroll
      for (int i = 0; i < 8; ++i)
        We[(size_t)(r0 + mb + i) * 4096 + nn + n] = f_to_bf16u(c[i]);
    }
  }
}

// ---------------- per-step kernels ----------------
__global__ void zero_kernel(float* __restrict__ p, int n) {
  int i = blockIdx.x * blockDim.x + threadIdx.x;
  if (i < n) p[i] = 0.0f;
}

// msg[e] = h[src[e]] @ We[e] ; scatter-add into agg[dst[e]]
// one wave per edge; lane owns output columns {2*lane, 2*lane+1}; packed-bf16 reads.
__global__ void msg_agg_kernel(const float* __restrict__ h,
                               const unsigned short* __restrict__ We,
                               const int* __restrict__ src, const int* __restrict__ dst,
                               float* __restrict__ agg) {
  int wave = threadIdx.x >> 5;
  int lane = threadIdx.x & 31;
  int e = blockIdx.x * 8 + wave;
  int s = src[e];
  int d = dst[e];
  float hv0 = h[(size_t)s * 64 + lane];
  float hv1 = h[(size_t)s * 64 + 32 + lane];
  const unsigned short* W = We + (size_t)e * 4096;
  __builtin_prefetch((const void*)(W + 2048), 0, 1);  // global_prefetch_b8
  float acc0 = 0.0f, acc1 = 0.0f;
#pragma unroll 8
  for (int i = 0; i < 32; ++i) {
    float hi = __shfl(hv0, i);
    unsigned int pk = *(const unsigned int*)(W + i * 64 + 2 * lane);
    acc0 += hi * __uint_as_float(pk << 16);
    acc1 += hi * __uint_as_float(pk & 0xffff0000u);
  }
#pragma unroll 8
  for (int i = 32; i < 64; ++i) {
    float hi = __shfl(hv1, i - 32);
    unsigned int pk = *(const unsigned int*)(W + i * 64 + 2 * lane);
    acc0 += hi * __uint_as_float(pk << 16);
    acc1 += hi * __uint_as_float(pk & 0xffff0000u);
  }
  atomicAdd(&agg[(size_t)d * 64 + 2 * lane], acc0);
  atomicAdd(&agg[(size_t)d * 64 + 2 * lane + 1], acc1);
}

// GRU cell (PyTorch gate order r,z,n); x = relu(agg + b_conv); in-place h update.
// 256 threads = 4 nodes x 64 lanes.
__global__ void gru_kernel(const float* __restrict__ agg, float* __restrict__ h,
                           const float* __restrict__ b_conv,
                           const float* __restrict__ w_ih, const float* __restrict__ w_hh,
                           const float* __restrict__ b_ih, const float* __restrict__ b_hh) {
  __shared__ float xs[4][64];
  __shared__ float hs[4][64];
  int t  = threadIdx.x & 63;
  int nl = threadIdx.x >> 6;
  int v  = blockIdx.x * 4 + nl;

  float x  = fmaxf(agg[(size_t)v * 64 + t] + b_conv[t], 0.0f);
  float hd = h[(size_t)v * 64 + t];
  xs[nl][t] = x;
  hs[nl][t] = hd;
  __syncthreads();

  float ir = b_ih[t],        hr = b_hh[t];
  float iz = b_ih[64 + t],   hz = b_hh[64 + t];
  float in_ = b_ih[128 + t], hn = b_hh[128 + t];
  const float* wr_i = w_ih + (size_t)t * 64;
  const float* wz_i = w_ih + (size_t)(64 + t) * 64;
  const float* wn_i = w_ih + (size_t)(128 + t) * 64;
  const float* wr_h = w_hh + (size_t)t * 64;
  const float* wz_h = w_hh + (size_t)(64 + t) * 64;
  const float* wn_h = w_hh + (size_t)(128 + t) * 64;
#pragma unroll 8
  for (int k = 0; k < 64; ++k) {
    float xk = xs[nl][k];
    float hk = hs[nl][k];
    ir += wr_i[k] * xk;  iz += wz_i[k] * xk;  in_ += wn_i[k] * xk;
    hr += wr_h[k] * hk;  hz += wz_h[k] * hk;  hn += wn_h[k] * hk;
  }
  float r = 1.0f / (1.0f + __expf(-(ir + hr)));
  float z = 1.0f / (1.0f + __expf(-(iz + hz)));
  float nn = tanhf(in_ + r * hn);
  h[(size_t)v * 64 + t] = (1.0f - z) * nn + z * hd;
}

// ---------------- workspace layout (bytes, 256-aligned) ----------------
static constexpr size_t OFF_WE  = 0;                                    // E*4096 bf16
static constexpr size_t OFF_HE  = OFF_WE  + (size_t)E_EDGES * 4096 * 2; // E*128 bf16
static constexpr size_t OFF_H   = OFF_HE  + (size_t)E_EDGES * 128 * 2;  // V*64 f32
static constexpr size_t OFF_AGG = OFF_H   + (size_t)V_NODES * 64 * 4;   // V*64 f32
static constexpr size_t OFF_PW2 = OFF_AGG + (size_t)V_NODES * 64 * 4;   // 4096x128 bf16 packed
static constexpr size_t OFF_PW1 = OFF_PW2 + (size_t)4096 * 128 * 2;     // 128x128 bf16 packed

extern "C" void kernel_launch(void* const* d_in, const int* in_sizes, int n_in,
                              void* d_out, int out_size, void* d_ws, size_t ws_size,
                              hipStream_t stream) {
  (void)in_sizes; (void)n_in; (void)out_size; (void)ws_size;

  const float* node_feats = (const float*)d_in[0];
  const float* edge_feats = (const float*)d_in[1];
  const int*   src        = (const int*)d_in[2];
  const int*   dst        = (const int*)d_in[3];
  const float* w_p1 = (const float*)d_in[4];
  const float* b_p1 = (const float*)d_in[5];
  const float* w_p2 = (const float*)d_in[6];
  const float* b_p2 = (const float*)d_in[7];
  const float* w_e1 = (const float*)d_in[8];
  const float* b_e1 = (const float*)d_in[9];
  const float* w_e2 = (const float*)d_in[10];
  const float* b_e2 = (const float*)d_in[11];
  const float* b_conv = (const float*)d_in[12];
  const float* w_ih = (const float*)d_in[13];
  const float* w_hh = (const float*)d_in[14];
  const float* b_ih = (const float*)d_in[15];
  const float* b_hh = (const float*)d_in[16];
  const float* w_d1 = (const float*)d_in[17];
  const float* b_d1 = (const float*)d_in[18];
  const float* w_d2 = (const float*)d_in[19];
  const float* b_d2 = (const float*)d_in[20];

  char* ws = (char*)d_ws;
  unsigned short* We = (unsigned short*)(ws + OFF_WE);
  unsigned short* He = (unsigned short*)(ws + OFF_HE);
  float* h   = (float*)(ws + OFF_H);
  float* agg = (float*)(ws + OFF_AGG);
  unsigned short* P2 = (unsigned short*)(ws + OFF_PW2);
  unsigned short* P1 = (unsigned short*)(ws + OFF_PW1);
  float* out = (float*)d_out;

  // pack edge-network weights to bf16 fragment-major layout (one-time, tiny)
  pack_b_kernel<<<(EDGE_HID / 16) * (EDGE_IN / 32), 32, 0, stream>>>(w_e1, EDGE_IN, P1);
  pack_b_kernel<<<(4096 / 16) * (EDGE_HID / 32), 32, 0, stream>>>(w_e2, EDGE_HID, P2);

  // h = project_node_feats(node_feats)       [WMMA bf16]
  fused2layer_kernel<NODE_IN><<<V_NODES / 16, 32, 0, stream>>>(
      node_feats, w_p1, b_p1, w_p2, b_p2, h);

  // He = relu(edge_feats @ w_e1^T + b_e1)    [WMMA bf16, packed B]
  edge_hidden_kernel<<<dim3(E_EDGES / 16, EDGE_HID / 64), 32, 0, stream>>>(
      edge_feats, P1, b_e1, He);

  // We = He @ w_e2^T + b_e2  (E x 4096, bf16) [WMMA bf16, 105 GFLOP, packed B]
  edge_we_kernel<<<dim3(E_EDGES / 16, 4096 / 512), 32, 0, stream>>>(
      He, P2, b_e2, We);

  // 9 message-passing steps (HBM-bound streaming of We)
  for (int step = 0; step < N_STEPS; ++step) {
    zero_kernel<<<(V_NODES * 64) / 256, 256, 0, stream>>>(agg, V_NODES * 64);
    msg_agg_kernel<<<E_EDGES / 8, 256, 0, stream>>>(h, We, src, dst, agg);
    gru_kernel<<<V_NODES / 4, 256, 0, stream>>>(agg, h, b_conv, w_ih, w_hh, b_ih, b_hh);
  }

  // out = decoder(h)                          [WMMA bf16]
  fused2layer_kernel<HDIM><<<V_NODES / 16, 32, 0, stream>>>(
      h, w_d1, b_d1, w_d2, b_d2, out);
}
